// Block_70970039599827
// MI455X (gfx1250) — matrix operand order repaired and checked by reading, not compiled, not run on hardware
//
#include <hip/hip_runtime.h>
#include <math.h>
#include <stdint.h>

typedef __bf16 bf16;
typedef __attribute__((ext_vector_type(16))) __bf16 v16bf;
typedef __attribute__((ext_vector_type(8)))  __bf16 v8bf;
typedef __attribute__((ext_vector_type(8)))  float  v8f;
typedef __attribute__((ext_vector_type(4)))  int    v4i_t;

#define BDIM 1024
#define SEQ  2048
#define NB   2
#define NH   16
#define DHD  64
#define DFF  4096
#define MROWS 4096   // B*S

__device__ inline v16bf pack16(v8bf lo, v8bf hi) {
  v16bf r;
#pragma unroll
  for (int i = 0; i < 8; ++i) { r[i] = lo[i]; r[i + 8] = hi[i]; }
  return r;
}

// 16-byte global -> LDS copy: async DMA on CDNA5 if available, else via VGPRs.
__device__ inline void copy16_g2l(const bf16* src, bf16* dst) {
#if __has_builtin(__builtin_amdgcn_global_load_async_to_lds_b128)
  __builtin_amdgcn_global_load_async_to_lds_b128(
      (__attribute__((address_space(1))) v4i_t*)(uintptr_t)src,
      (__attribute__((address_space(3))) v4i_t*)(uint32_t)(uintptr_t)dst,
      0, 0);
#else
  *(v8bf*)dst = *(const v8bf*)src;
#endif
}

__device__ inline void wait_async_all() {
#if __has_builtin(__builtin_amdgcn_s_wait_asynccnt)
  __builtin_amdgcn_s_wait_asynccnt(0);
#elif __has_builtin(__builtin_amdgcn_global_load_async_to_lds_b128)
  asm volatile("s_wait_asynccnt 0" ::: "memory");
#endif
}

// ---------------------------------------------------------------------------
// Weight fp32 (K x N) -> bf16 transposed (N x K)
// ---------------------------------------------------------------------------
__global__ void convT_kernel(const float* __restrict__ W, bf16* __restrict__ Wt,
                             int K, int N) {
  int idx = blockIdx.x * 256 + threadIdx.x;
  if (idx < K * N) {
    int k = idx / N, n = idx % N;
    Wt[(size_t)n * K + k] = (bf16)W[idx];
  }
}

// ---------------------------------------------------------------------------
// LayerNorm: fp32 row -> bf16 row   (one block per row, 256 threads, D=1024)
// ---------------------------------------------------------------------------
__global__ __launch_bounds__(256)
void ln_kernel(const float* __restrict__ x, const float* __restrict__ g,
               const float* __restrict__ bb, bf16* __restrict__ out) {
  const int row = blockIdx.x;
  const int tid = threadIdx.x;
  const float* xr = x + (size_t)row * BDIM;
  __shared__ float red[256], red2[256];
  float s = 0.f, s2 = 0.f;
#pragma unroll
  for (int i = tid; i < BDIM; i += 256) { float v = xr[i]; s += v; s2 += v * v; }
  red[tid] = s; red2[tid] = s2;
  __syncthreads();
  for (int off = 128; off > 0; off >>= 1) {
    if (tid < off) { red[tid] += red[tid + off]; red2[tid] += red2[tid + off]; }
    __syncthreads();
  }
  float mu  = red[0] * (1.f / BDIM);
  float var = red2[0] * (1.f / BDIM) - mu * mu;
  float rs  = rsqrtf(var + 1e-5f);
#pragma unroll
  for (int i = tid; i < BDIM; i += 256) {
    out[(size_t)row * BDIM + i] = (bf16)((xr[i] - mu) * rs * g[i] + bb[i]);
  }
}

// ---------------------------------------------------------------------------
// GEMM: C[M,N] = A[M,K](bf16) * Wt[N,K]^T(bf16) + bias, epilogue options.
// block = 256 threads = 8 waves; wave tile = 16 rows x 64 cols (4 WMMA N-tiles)
// B tile (64 rows of Wt x 64 k) staged in LDS via async DMA, double-buffered.
// grid = (N/64, M/128)
// ---------------------------------------------------------------------------
#define BKT 64

__global__ __launch_bounds__(256)
void gemm_wmma_kernel(const bf16* __restrict__ A, const bf16* __restrict__ Wt,
                      const float* __restrict__ bias,
                      const float* __restrict__ resid,
                      float* __restrict__ outF, bf16* __restrict__ outB,
                      int M, int N, int K, int dogelu) {
  __shared__ __align__(16) bf16 Bt[2][64 * BKT];   // 2 x 8 KB

  const int tid    = threadIdx.x;
  const int lane   = tid & 31;
  const int wave   = tid >> 5;
  const int laneHi = lane >> 4;
  const int lanelo = lane & 15;
  const int m0 = blockIdx.y * 128 + wave * 16;
  const int n0 = blockIdx.x * 64;

  const bf16* arow = A + (size_t)(m0 + lanelo) * K;

  // stage one 64x64(bf16) B tile: 512 x 16B segments, 2 per thread
  auto issue_tile = [&](int buf, int kbase) {
#pragma unroll
    for (int j = 0; j < 2; ++j) {
      int s = tid * 2 + j;
      int row = s >> 3;            // 0..63  (Wt row within block tile)
      int ko  = (s & 7) * 8;       // 0..56  (k offset in halfs)
      copy16_g2l(Wt + (size_t)(n0 + row) * K + kbase + ko,
                 &Bt[buf][row * BKT + ko]);
    }
  };

  v8f acc[4];
#pragma unroll
  for (int t = 0; t < 4; ++t)
#pragma unroll
    for (int r = 0; r < 8; ++r) acc[t][r] = 0.f;

  const int nTiles = K / BKT;
  issue_tile(0, 0);

  for (int i = 0; i < nTiles; ++i) {
    wait_async_all();
    __syncthreads();
    if (i + 1 < nTiles) issue_tile((i + 1) & 1, (i + 1) * BKT);

    const bf16* bbase = &Bt[i & 1][0];
#pragma unroll
    for (int ks = 0; ks < 2; ++ks) {
      const int k = i * BKT + ks * 32;
      v8bf alo = *(const v8bf*)(arow + k + laneHi * 8);
      v8bf ahi = *(const v8bf*)(arow + k + 16 + laneHi * 8);
      v16bf afrag = pack16(alo, ahi);
#pragma unroll
      for (int t = 0; t < 4; ++t) {
        const bf16* brow = bbase + (t * 16 + lanelo) * BKT + ks * 32;
        v16bf bfrag = pack16(*(const v8bf*)(brow + laneHi * 8),
                             *(const v8bf*)(brow + 16 + laneHi * 8));
        acc[t] = __builtin_amdgcn_wmma_f32_16x16x32_bf16(
            false, afrag, false, bfrag, (short)0, acc[t], false, false);
      }
    }
  }

#pragma unroll
  for (int t = 0; t < 4; ++t) {
#pragma unroll
    for (int r = 0; r < 8; ++r) {
      int row = m0 + r + 8 * laneHi;
      int col = n0 + t * 16 + lanelo;
      size_t off = (size_t)row * N + col;
      float v = acc[t][r] + bias[col];
      if (resid) v += resid[off];
      if (dogelu) v = 0.5f * v * (1.f + erff(v * 0.70710678f));
      if (outF) outF[off] = v;
      if (outB) outB[off] = (bf16)v;
    }
  }
}

// ---------------------------------------------------------------------------
// Flash attention, causal. block = 128 threads = 4 waves, each wave owns 16
// query rows; keys streamed in chunks of 32. grid = (S/64, B*H)
// ---------------------------------------------------------------------------
#define CK 32

__global__ __launch_bounds__(128)
void attn_kernel(const bf16* __restrict__ Q, const bf16* __restrict__ Kb,
                 const bf16* __restrict__ V, bf16* __restrict__ O) {
  const int lane   = threadIdx.x & 31;
  const int wave   = threadIdx.x >> 5;
  const int laneHi = lane >> 4;
  const int lanelo = lane & 15;
  const int qblk = blockIdx.x;
  const int bh   = blockIdx.y;
  const int b = bh / NH, h = bh % NH;
  const int q0 = qblk * 64 + wave * 16;
  const float scale = 0.125f;  // 1/sqrt(64)

  __shared__ __align__(16) bf16 Vt[DHD * CK];        // [dh][key] transposed
  __shared__ __align__(16) bf16 PL[4][16 * CK];      // per-wave P tile

  // persistent Q fragments (K-dim = DH = 64 -> two 32-wide steps)
  const bf16* qptr = Q + ((size_t)(b * SEQ + q0 + lanelo)) * BDIM + h * DHD;
  v16bf qa[2];
#pragma unroll
  for (int ks = 0; ks < 2; ++ks) {
    v8bf lo = *(const v8bf*)(qptr + ks * 32 + laneHi * 8);
    v8bf hi = *(const v8bf*)(qptr + ks * 32 + 16 + laneHi * 8);
    qa[ks] = pack16(lo, hi);
  }

  v8f o[4];
#pragma unroll
  for (int t = 0; t < 4; ++t)
#pragma unroll
    for (int r = 0; r < 8; ++r) o[t][r] = 0.f;
  float mrow[8], lrow[8];
#pragma unroll
  for (int r = 0; r < 8; ++r) { mrow[r] = -1e30f; lrow[r] = 0.f; }

  const int nchunks = qblk * 2 + 2;
  for (int c = 0; c < nchunks; ++c) {
    const int c0 = c * CK;
    // stage V chunk transposed into LDS (32 keys x 64 dh)
    for (int j = 0; j < 16; ++j) {
      int idx = threadIdx.x * 16 + j;
      int key = idx >> 6, dh = idx & 63;
      Vt[dh * CK + key] =
          V[((size_t)(b * SEQ + c0 + key)) * BDIM + h * DHD + dh];
    }
    __syncthreads();

    // scores: two 16-key sub-tiles, each accumulated over DH=64
    v8f s[2];
#pragma unroll
    for (int t = 0; t < 2; ++t) {
#pragma unroll
      for (int r = 0; r < 8; ++r) s[t][r] = 0.f;
      const bf16* kptr =
          Kb + ((size_t)(b * SEQ + c0 + t * 16 + lanelo)) * BDIM + h * DHD;
#pragma unroll
      for (int ks = 0; ks < 2; ++ks) {
        v8bf lo = *(const v8bf*)(kptr + ks * 32 + laneHi * 8);
        v8bf hi = *(const v8bf*)(kptr + ks * 32 + 16 + laneHi * 8);
        v16bf kfrag = pack16(lo, hi);
        s[t] = __builtin_amdgcn_wmma_f32_16x16x32_bf16(
            false, qa[ks], false, kfrag, (short)0, s[t], false, false);
      }
    }

    // online softmax per row (row lives in 16 lanes of one half-wave)
#pragma unroll
    for (int r = 0; r < 8; ++r) {
      int qg = q0 + r + 8 * laneHi;
      float v0 = s[0][r] * scale;
      float v1 = s[1][r] * scale;
      if (c0 + lanelo > qg)      v0 = -1e30f;
      if (c0 + 16 + lanelo > qg) v1 = -1e30f;
      float mx = fmaxf(v0, v1);
#pragma unroll
      for (int msk = 8; msk >= 1; msk >>= 1) mx = fmaxf(mx, __shfl_xor(mx, msk, 32));
      float mnew = fmaxf(mrow[r], mx);
      float alpha = __expf(mrow[r] - mnew);
      mrow[r] = mnew;
      float p0 = __expf(v0 - mnew);
      float p1 = __expf(v1 - mnew);
      float rs = p0 + p1;
#pragma unroll
      for (int msk = 8; msk >= 1; msk >>= 1) rs += __shfl_xor(rs, msk, 32);
      lrow[r] = lrow[r] * alpha + rs;
#pragma unroll
      for (int t = 0; t < 4; ++t) o[t][r] *= alpha;
      PL[wave][(r + 8 * laneHi) * CK + lanelo]      = (bf16)p0;
      PL[wave][(r + 8 * laneHi) * CK + 16 + lanelo] = (bf16)p1;
    }
    asm volatile("s_wait_dscnt 0" ::: "memory");

    // P (16x32) as A fragment from per-wave LDS
    const bf16* plrow = &PL[wave][lanelo * CK];
    v16bf pa = pack16(*(const v8bf*)(plrow + laneHi * 8),
                      *(const v8bf*)(plrow + 16 + laneHi * 8));
    // O += P * V  (4 dh tiles, K = 32 keys)
#pragma unroll
    for (int t = 0; t < 4; ++t) {
      const bf16* vtrow = &Vt[(t * 16 + lanelo) * CK];
      v16bf vfrag = pack16(*(const v8bf*)(vtrow + laneHi * 8),
                           *(const v8bf*)(vtrow + 16 + laneHi * 8));
      o[t] = __builtin_amdgcn_wmma_f32_16x16x32_bf16(
          false, pa, false, vfrag, (short)0, o[t], false, false);
    }
    __syncthreads();
  }

  // normalize and write bf16 output [B*S, D] at head offset
#pragma unroll
  for (int r = 0; r < 8; ++r) {
    float inv = 1.f / lrow[r];
    int qg = q0 + r + 8 * laneHi;
#pragma unroll
    for (int t = 0; t < 4; ++t) {
      O[((size_t)(b * SEQ + qg)) * BDIM + h * DHD + t * 16 + lanelo] =
          (bf16)(o[t][r] * inv);
    }
  }
}

// ---------------------------------------------------------------------------
extern "C" void kernel_launch(void* const* d_in, const int* in_sizes, int n_in,
                              void* d_out, int out_size, void* d_ws, size_t ws_size,
                              hipStream_t stream) {
  const float* x     = (const float*)d_in[0];
  const float* ln1_g = (const float*)d_in[1];
  const float* ln1_b = (const float*)d_in[2];
  const float* Wq    = (const float*)d_in[3];
  const float* bq    = (const float*)d_in[4];
  const float* Wk    = (const float*)d_in[5];
  const float* bk    = (const float*)d_in[6];
  const float* Wv    = (const float*)d_in[7];
  const float* bv    = (const float*)d_in[8];
  const float* Wo    = (const float*)d_in[9];
  const float* bo    = (const float*)d_in[10];
  const float* ln2_g = (const float*)d_in[11];
  const float* ln2_b = (const float*)d_in[12];
  const float* W1    = (const float*)d_in[13];
  const float* b1    = (const float*)d_in[14];
  const float* W2    = (const float*)d_in[15];
  const float* b2    = (const float*)d_in[16];

  char* w = (char*)d_ws;
  auto take = [&](size_t bytes) { char* p = w; w += (bytes + 255) & ~(size_t)255; return p; };

  bf16* hb    = (bf16*)take((size_t)MROWS * BDIM * 2);
  bf16* qb    = (bf16*)take((size_t)MROWS * BDIM * 2);
  bf16* kb    = (bf16*)take((size_t)MROWS * BDIM * 2);
  bf16* vb    = (bf16*)take((size_t)MROWS * BDIM * 2);
  bf16* attnb = (bf16*)take((size_t)MROWS * BDIM * 2);
  bf16* h2b   = (bf16*)take((size_t)MROWS * BDIM * 2);
  bf16* ffb   = (bf16*)take((size_t)MROWS * DFF * 2);
  float* x2   = (float*)take((size_t)MROWS * BDIM * 4);
  bf16* WqT   = (bf16*)take((size_t)BDIM * BDIM * 2);
  bf16* WkT   = (bf16*)take((size_t)BDIM * BDIM * 2);
  bf16* WvT   = (bf16*)take((size_t)BDIM * BDIM * 2);
  bf16* WoT   = (bf16*)take((size_t)BDIM * BDIM * 2);
  bf16* W1T   = (bf16*)take((size_t)BDIM * DFF * 2);
  bf16* W2T   = (bf16*)take((size_t)DFF * BDIM * 2);

  const int cvt1 = (BDIM * BDIM + 255) / 256;
  const int cvt2 = (BDIM * DFF + 255) / 256;
  convT_kernel<<<cvt1, 256, 0, stream>>>(Wq, WqT, BDIM, BDIM);
  convT_kernel<<<cvt1, 256, 0, stream>>>(Wk, WkT, BDIM, BDIM);
  convT_kernel<<<cvt1, 256, 0, stream>>>(Wv, WvT, BDIM, BDIM);
  convT_kernel<<<cvt1, 256, 0, stream>>>(Wo, WoT, BDIM, BDIM);
  convT_kernel<<<cvt2, 256, 0, stream>>>(W1, W1T, BDIM, DFF);
  convT_kernel<<<cvt2, 256, 0, stream>>>(W2, W2T, DFF, BDIM);

  // LN1 -> bf16
  ln_kernel<<<MROWS, 256, 0, stream>>>(x, ln1_g, ln1_b, hb);

  // Q/K/V projections (bf16 outputs)
  dim3 gD(BDIM / 64, MROWS / 128);
  gemm_wmma_kernel<<<gD, 256, 0, stream>>>(hb, WqT, bq, nullptr, nullptr, qb,
                                           MROWS, BDIM, BDIM, 0);
  gemm_wmma_kernel<<<gD, 256, 0, stream>>>(hb, WkT, bk, nullptr, nullptr, kb,
                                           MROWS, BDIM, BDIM, 0);
  gemm_wmma_kernel<<<gD, 256, 0, stream>>>(hb, WvT, bv, nullptr, nullptr, vb,
                                           MROWS, BDIM, BDIM, 0);

  // flash attention
  dim3 gA(SEQ / 64, NB * NH);
  attn_kernel<<<gA, 128, 0, stream>>>(qb, kb, vb, attnb);

  // output projection + residual -> x2 (fp32)
  gemm_wmma_kernel<<<gD, 256, 0, stream>>>(attnb, WoT, bo, x, x2, nullptr,
                                           MROWS, BDIM, BDIM, 0);

  // LN2 -> bf16
  ln_kernel<<<MROWS, 256, 0, stream>>>(x2, ln2_g, ln2_b, h2b);

  // FFN1 + GELU -> bf16
  dim3 gF1(DFF / 64, MROWS / 128);
  gemm_wmma_kernel<<<gF1, 256, 0, stream>>>(h2b, W1T, b1, nullptr, nullptr, ffb,
                                            MROWS, DFF, BDIM, 1);

  // FFN2 + residual -> d_out (fp32)
  gemm_wmma_kernel<<<gD, 256, 0, stream>>>(ffb, W2T, b2, x2, (float*)d_out,
                                           nullptr, MROWS, BDIM, DFF, 0);
}
